// nnvit_82042465288319
// MI455X (gfx1250) — matrix operand order repaired and checked by reading, not compiled, not run on hardware
//
#include <hip/hip_runtime.h>
#include <hip/hip_bf16.h>

typedef __attribute__((ext_vector_type(16))) _Float16 v16h;
typedef __attribute__((ext_vector_type(8)))  _Float16 v8h;
typedef __attribute__((ext_vector_type(8)))  float    v8f;
typedef __attribute__((ext_vector_type(4)))  unsigned int v4u;
typedef __attribute__((ext_vector_type(8)))  int      v8i;
typedef __attribute__((ext_vector_type(4)))  int      v4i;

#define MB    32   // batch elements per block (2 WMMA M-tiles)
#define NTOK  5
#define TDIM  32
#define NHEAD 8
#define HDIM  4
#define DEEP  2

#if defined(__HIP_DEVICE_COMPILE__) && __has_builtin(__builtin_amdgcn_tensor_load_to_lds)
#define USE_TDM 1
#else
#define USE_TDM 0
#endif

__global__ __launch_bounds__(256) void nnvit_82042465288319_kernel(
    const float* __restrict__ x,     const float* __restrict__ token,
    const float* __restrict__ qkvw,  const float* __restrict__ qkvb,
    const float* __restrict__ projw, const float* __restrict__ projb,
    const float* __restrict__ posb,  float* __restrict__ out)
{
  __shared__ __align__(16) _Float16 s_qkvw[DEEP][96][TDIM];     // 12 KB f16 weights
  __shared__ __align__(16) _Float16 s_projw[DEEP][TDIM][TDIM];  // 4 KB
  __shared__ float    s_qkvb[DEEP][96];
  __shared__ float    s_projb[DEEP][TDIM];
  __shared__ float    s_posb[DEEP][NHEAD][NTOK*NTOK];
  __shared__ __align__(16) float    s_Y [MB][NTOK][TDIM];       // 20 KB fp32 master (residual)
  __shared__ __align__(16) _Float16 s_Yh[MB][NTOK][TDIM];       // 10 KB f16 shadow (WMMA A)
  __shared__ __align__(16) _Float16 s_qkv[MB][NTOK][96];        // 30 KB
  __shared__ __align__(16) _Float16 s_o [MB][NTOK][TDIM];       // 10 KB

  const int tid  = threadIdx.x;
  const int lane = tid & 31;
  // Scalarized wave index: job loops become SALU-controlled, EXEC stays all-1s
  // around every WMMA (ISA 7.12 requirement) by construction.
  const int wave = __builtin_amdgcn_readfirstlane(tid >> 5);
  const int lo   = lane & 15;   // N / M-row within tile
  const int g    = lane >> 4;   // lane group (K-half select)
  const int b0   = blockIdx.x * MB;

#if USE_TDM
  // ---- TDM: DMA the 32x128f32 x-tile into s_Y rows 1..4, striding over the
  //      token rows via LDS padding (interval=128 DW -> pad 32 DW = one token row)
  if (wave == 0) {
    unsigned int lds_off = (unsigned int)(uintptr_t)&s_Y[0][1][0];
    unsigned long long ga = (unsigned long long)(uintptr_t)(x + (size_t)b0 * 128);
    v4u g0;
    g0[0] = 1u;                                          // count=1 (valid), user flags 0
    g0[1] = lds_off;                                     // lds_addr (bytes)
    g0[2] = (unsigned int)(ga & 0xffffffffu);            // global_addr[31:0]
    g0[3] = (unsigned int)((ga >> 32) & 0x1ffffffu) | (2u << 30); // addr[56:32] | type=2
    v8i g1;
    g1[0] = (int)((2u << 16) | (1u << 20) | (6u << 22) | (31u << 25));
            // data_size=4B | pad_enable | pad_interval=128DW | pad_amount=32DW
    g1[1] = (int)(128u << 16);   // tensor_dim0 = 128 (elements)
    g1[2] = (int)(32u  << 16);   // tensor_dim1 = 32
    g1[3] = (int)(128u << 16);   // tile_dim0   = 128
    g1[4] = (int)(32u);          // tile_dim1   = 32, tile_dim2 = 0
    g1[5] = (int)(128u);         // tensor_dim0_stride = 128
    g1[6] = 0; g1[7] = 0;
    v4i gz; gz[0]=0; gz[1]=0; gz[2]=0; gz[3]=0;
#if __clang_major__ >= 23
    v8i gz8; gz8[0]=0; gz8[1]=0; gz8[2]=0; gz8[3]=0; gz8[4]=0; gz8[5]=0; gz8[6]=0; gz8[7]=0;
    __builtin_amdgcn_tensor_load_to_lds(g0, g1, gz, gz, gz8, 0);
#else
    __builtin_amdgcn_tensor_load_to_lds(g0, g1, gz, gz, 0);
#endif
  }
#endif

  // ---- stage weights (f32 -> f16) + biases into LDS (overlaps with TDM) ----
  for (int t = tid; t < DEEP*96*TDIM;        t += 256) ((_Float16*)s_qkvw)[t]  = (_Float16)qkvw[t];
  for (int t = tid; t < DEEP*TDIM*TDIM;      t += 256) ((_Float16*)s_projw)[t] = (_Float16)projw[t];
  for (int t = tid; t < DEEP*96;             t += 256) ((float*)s_qkvb)[t]  = qkvb[t];
  for (int t = tid; t < DEEP*TDIM;           t += 256) ((float*)s_projb)[t] = projb[t];
  for (int t = tid; t < DEEP*NHEAD*NTOK*NTOK;t += 256) ((float*)s_posb)[t]  = posb[t];

  // ---- token row ----
  for (int t = tid; t < MB*TDIM; t += 256) s_Y[t >> 5][0][t & 31] = token[t & 31];

#if USE_TDM
  __builtin_amdgcn_s_wait_tensorcnt(0);   // only wave 0 has TENSORcnt>0; harmless on others
#else
  {
    const float4* xv = (const float4*)(x + (size_t)b0 * 128);
    for (int t = tid; t < MB*32; t += 256) {
      int b = t >> 5, j4 = t & 31;
      float4 val = xv[b*32 + j4];
      float4* dst = (float4*)(&s_Y[b][0][0] + TDIM);
      dst[j4] = val;
    }
  }
#endif
  __syncthreads();

  // ---- f16 shadow of Y (one conversion, reused by all 6 N-tiles of GEMM1) ----
  for (int t = tid; t < MB*NTOK*TDIM; t += 256)
    ((_Float16*)s_Yh)[t] = (_Float16)(((const float*)s_Y)[t]);
  __syncthreads();

  for (int layer = 0; layer < DEEP; ++layer) {
    // ======== GEMM1: qkv = Y @ Wqkv^T + b  (v_wmma_f32_16x16x32_f16, K==TD==32) ========
    for (int job = wave; job < (MB/16)*NTOK*6; job += 8) {   // scalar loop: EXEC all-1s
      int mt  = job / (NTOK*6);
      int rem = job % (NTOK*6);
      int n = rem / 6, jt = rem % 6;
      int bl  = mt*16 + lo;       // A: M = lane%16
      int col = jt*16 + lo;       // B/C/D: N = lane%16
      // A fragment: halves K=8g..8g+7 and K=16+8g..16+8g+7  (2x ds_load_b128)
      v8h alo = ((const v8h*)&s_Yh[bl][n][0])[g];
      v8h ahi = ((const v8h*)&s_Yh[bl][n][16])[g];
      v16h a = __builtin_shufflevector(alo, ahi, 0,1,2,3,4,5,6,7,8,9,10,11,12,13,14,15);
      // B fragment: halves K=8g..8g+7 and K=16+8g..16+8g+7 of weight row `col`  (2x ds_load_b128)
      v8h blo = ((const v8h*)&s_qkvw[layer][col][0])[g];
      v8h bhi = ((const v8h*)&s_qkvw[layer][col][16])[g];
      v16h bf = __builtin_shufflevector(blo, bhi, 0,1,2,3,4,5,6,7,8,9,10,11,12,13,14,15);
      float bias = s_qkvb[layer][col];
      v8f c;
      #pragma unroll
      for (int r = 0; r < 8; ++r) c[r] = bias;               // fold bias into C
      v8f d = __builtin_amdgcn_wmma_f32_16x16x32_f16(false, a, false, bf,
                                                     (short)0, c, false, false);
      #pragma unroll
      for (int r = 0; r < 8; ++r)                            // D: M = r + 8*g
        s_qkv[mt*16 + 8*g + r][n][col] = (_Float16)d[r];
    }
    __syncthreads();

    // ======== attention: one (batch, head) pair per thread (256 = 32*8) ========
    {
      int b = tid >> 3;
      int h = tid & 7;
      float q[NTOK][HDIM], kk[NTOK][HDIM], vv[NTOK][HDIM];
      #pragma unroll
      for (int n = 0; n < NTOK; ++n)
        #pragma unroll
        for (int dd = 0; dd < HDIM; ++dd) {
          q[n][dd]  = (float)s_qkv[b][n][      h*HDIM + dd];
          kk[n][dd] = (float)s_qkv[b][n][32  + h*HDIM + dd];
          vv[n][dd] = (float)s_qkv[b][n][64  + h*HDIM + dd];
        }
      #pragma unroll
      for (int n = 0; n < NTOK; ++n) {
        float sc[NTOK], mx = -3.0e38f;
        #pragma unroll
        for (int m = 0; m < NTOK; ++m) {
          float s = 0.f;
          #pragma unroll
          for (int dd = 0; dd < HDIM; ++dd) s += q[n][dd]*kk[m][dd];
          s = s*0.5f + s_posb[layer][h][n*NTOK + m];         // scale = HD^-0.5 = 0.5
          sc[m] = s;
          mx = fmaxf(mx, s);
        }
        float sum = 0.f;
        #pragma unroll
        for (int m = 0; m < NTOK; ++m) { sc[m] = __expf(sc[m] - mx); sum += sc[m]; }
        float inv = 1.f / sum;
        #pragma unroll
        for (int dd = 0; dd < HDIM; ++dd) {
          float o = 0.f;
          #pragma unroll
          for (int m = 0; m < NTOK; ++m) o += sc[m]*vv[m][dd];
          s_o[b][n][h*HDIM + dd] = (_Float16)(o * inv);
        }
      }
    }
    __syncthreads();

    // ======== GEMM2 + residual: Y += O @ Wproj^T + b ========
    for (int job = wave; job < (MB/16)*NTOK*2; job += 8) {
      int mt  = job / (NTOK*2);
      int rem = job % (NTOK*2);
      int n = rem / 2, jt = rem % 2;
      int bl  = mt*16 + lo;
      int col = jt*16 + lo;
      v8h alo = ((const v8h*)&s_o[bl][n][0])[g];
      v8h ahi = ((const v8h*)&s_o[bl][n][16])[g];
      v16h a = __builtin_shufflevector(alo, ahi, 0,1,2,3,4,5,6,7,8,9,10,11,12,13,14,15);
      v8h blo = ((const v8h*)&s_projw[layer][col][0])[g];
      v8h bhi = ((const v8h*)&s_projw[layer][col][16])[g];
      v16h bf = __builtin_shufflevector(blo, bhi, 0,1,2,3,4,5,6,7,8,9,10,11,12,13,14,15);
      float bias = s_projb[layer][col];
      v8f c;
      #pragma unroll
      for (int r = 0; r < 8; ++r) c[r] = bias;
      v8f d = __builtin_amdgcn_wmma_f32_16x16x32_f16(false, a, false, bf,
                                                     (short)0, c, false, false);
      #pragma unroll
      for (int r = 0; r < 8; ++r) {                          // disjoint (row,n,col): no races
        int row = mt*16 + 8*g + r;
        float yv = s_Y[row][n][col] + d[r];
        s_Y[row][n][col]  = yv;
        s_Yh[row][n][col] = (_Float16)yv;                    // keep f16 shadow current
      }
    }
    __syncthreads();
  }

  // ---- emit y[:, 0, :] ----
  for (int t = tid; t < MB*TDIM; t += 256)
    out[(size_t)(b0 + (t >> 5))*TDIM + (t & 31)] = s_Y[t >> 5][0][t & 31];
}

extern "C" void kernel_launch(void* const* d_in, const int* in_sizes, int n_in,
                              void* d_out, int out_size, void* d_ws, size_t ws_size,
                              hipStream_t stream) {
  const float* x     = (const float*)d_in[0];
  const float* token = (const float*)d_in[1];
  const float* qkvw  = (const float*)d_in[2];
  const float* qkvb  = (const float*)d_in[3];
  const float* projw = (const float*)d_in[4];
  const float* projb = (const float*)d_in[5];
  const float* posb  = (const float*)d_in[6];
  float* out = (float*)d_out;

  int B = in_sizes[0] / 128;          // 262144
  dim3 grid(B / MB);                  // 8192 blocks of 256 threads (8 wave32)
  nnvit_82042465288319_kernel<<<grid, 256, 0, stream>>>(
      x, token, qkvw, qkvb, projw, projb, posb, out);
}
